// DynamicFocus_6270652252931
// MI455X (gfx1250) — matrix operand order repaired and checked by reading, not compiled, not run on hardware
//
#include <hip/hip_runtime.h>
#include <hip/hip_bf16.h>
#include <stdint.h>

// ---- problem constants (match reference) ----
#define T_SZ 2048
#define D_SZ 1024
#define NH   16
#define HD   64

typedef __attribute__((ext_vector_type(16))) _Float16 v16h;
typedef __attribute__((ext_vector_type(8)))  _Float16 v8h;
typedef __attribute__((ext_vector_type(8)))  float    v8f;
typedef __attribute__((ext_vector_type(4)))  float    v4f;

__device__ __forceinline__ v8f zero_v8f() {
    v8f z;
    for (int i = 0; i < 8; i++) z[i] = 0.0f;
    return z;
}

// Build a 16-half fragment from two 8-half LDS chunks (16B aligned each).
__device__ __forceinline__ v16h frag2(const _Float16* p0, const _Float16* p1) {
    v8h a = *(const v8h*)p0;
    v8h b = *(const v8h*)p1;
    return __builtin_shufflevector(a, b, 0,1,2,3,4,5,6,7,8,9,10,11,12,13,14,15);
}

__device__ __forceinline__ v8f wmma16(v16h a, v16h b, v8f c) {
    // D = A*B + C, fp32 accumulate (codegen-verified builtin)
    return __builtin_amdgcn_wmma_f32_16x16x32_f16(false, a, false, b,
                                                  (short)0, c, false, false);
}

// Per-lane 16B async DMA: global -> LDS, tracked by ASYNCcnt.
__device__ __forceinline__ void async_copy_b128(void* lds_dst, const void* gsrc) {
    unsigned lds = (unsigned)(uintptr_t)lds_dst;   // low 32 bits = LDS byte offset
    unsigned long long ga = (unsigned long long)(uintptr_t)gsrc;
    asm volatile("global_load_async_to_lds_b128 %0, %1, off"
                 :: "v"(lds), "v"(ga) : "memory");
}
__device__ __forceinline__ void wait_async() {
    asm volatile("s_wait_asynccnt 0x0" ::: "memory");
}

// =====================================================================
// Kernel 1: projection GEMM  out_f16[M,N] = f16( X_f32[M,K] @ W_f32[K,N] + b )
// M=T, N=K=D.  BM=128 BN=128 BK=32, 8 waves, each wave 32x64 (2x4 tiles).
// =====================================================================
__global__ __launch_bounds__(256) void proj_kernel(
    const float* __restrict__ X, const float* __restrict__ W,
    const float* __restrict__ bias, _Float16* __restrict__ out)
{
    const int BM = 128, BN = 128, BK = 32, LDB = BK + 8; // LDB row = 80B (16B mult.)
    __shared__ _Float16 As[BM * BK];     // row-major [m][k], 8KB
    __shared__ _Float16 Bs[BN * LDB];    // transposed [n][k], 10KB

    const int tid = threadIdx.x, lane = tid & 31, wid = tid >> 5;
    const int bm = blockIdx.y * BM, bn = blockIdx.x * BN;
    const int wm = (wid >> 1) * 32, wn = (wid & 1) * 64;

    v8f acc[2][4];
    for (int i = 0; i < 2; i++) for (int j = 0; j < 4; j++) acc[i][j] = zero_v8f();

    const int ar0 = tid >> 3, ac4 = (tid & 7) * 4;   // A staging coords
    const int bk0 = tid >> 5, bc4 = (tid & 31) * 4;  // B staging coords

    for (int k0 = 0; k0 < D_SZ; k0 += BK) {
        __syncthreads();
        // stage A (128x32 f32 -> f16, row-major): 4 batched loads per thread
        v4f ax[4];
#pragma unroll
        for (int it = 0; it < 4; it++)
            ax[it] = *(const v4f*)(X + (size_t)(bm + ar0 + it * 32) * D_SZ + k0 + ac4);
        // stage B (W [k][n] f32 -> LDS [n][k] f16): 4 batched loads per thread
        v4f bx[4];
#pragma unroll
        for (int it = 0; it < 4; it++)
            bx[it] = *(const v4f*)(W + (size_t)(k0 + bk0 + it * 8) * D_SZ + bn + bc4);
#pragma unroll
        for (int it = 0; it < 4; it++) {
            _Float16* d = As + (ar0 + it * 32) * BK + ac4;
            d[0] = (_Float16)ax[it][0]; d[1] = (_Float16)ax[it][1];
            d[2] = (_Float16)ax[it][2]; d[3] = (_Float16)ax[it][3];
        }
#pragma unroll
        for (int it = 0; it < 4; it++) {
#pragma unroll
            for (int j = 0; j < 4; j++)
                Bs[(bc4 + j) * LDB + (bk0 + it * 8)] = (_Float16)bx[it][j];
        }
        if (k0 + BK < D_SZ) {
            __builtin_prefetch(X + (size_t)(bm + ar0) * D_SZ + k0 + BK, 0, 1);
            __builtin_prefetch(W + (size_t)(k0 + BK + bk0) * D_SZ + bn, 0, 1);
        }
        __syncthreads();
#pragma unroll
        for (int mi = 0; mi < 2; mi++) {
            const _Float16* ap = As + (wm + mi * 16 + (lane & 15)) * BK + (lane >> 4) * 8;
            v16h af = frag2(ap, ap + 16);
#pragma unroll
            for (int ni = 0; ni < 4; ni++) {
                const _Float16* bp = Bs + (wn + ni * 16 + (lane & 15)) * LDB + (lane >> 4) * 8;
                v16h bf = frag2(bp, bp + 16);
                acc[mi][ni] = wmma16(af, bf, acc[mi][ni]);
            }
        }
    }
#pragma unroll
    for (int mi = 0; mi < 2; mi++)
#pragma unroll
        for (int ni = 0; ni < 4; ni++) {
            const int Mb = bm + wm + mi * 16 + (lane >> 4) * 8;
            const int N  = bn + wn + ni * 16 + (lane & 15);
            const float bv = bias[N];
#pragma unroll
            for (int r = 0; r < 8; r++)
                out[(size_t)(Mb + r) * D_SZ + N] = (_Float16)(acc[mi][ni][r] + bv);
        }
}

// =====================================================================
// Kernel 2: per-(head,row) squared norms of qp/kp (from the f16 values)
// =====================================================================
__global__ __launch_bounds__(256) void norm_kernel(
    const _Float16* __restrict__ qp, const _Float16* __restrict__ kp,
    float* __restrict__ q2, float* __restrict__ k2)
{
    int idx = blockIdx.x * 256 + threadIdx.x;        // 0 .. 2*T*NH-1
    const int which = idx >= T_SZ * NH;
    int i = which ? idx - T_SZ * NH : idx;
    const int h = i & (NH - 1);
    const int t = i >> 4;
    const _Float16* p = (which ? kp : qp) + (size_t)t * D_SZ + h * HD;
    float s = 0.0f;
#pragma unroll
    for (int c = 0; c < HD; c += 8) {
        v8h x = *(const v8h*)(p + c);
#pragma unroll
        for (int j = 0; j < 8; j++) { float f = (float)x[j]; s += f * f; }
    }
    (which ? k2 : q2)[h * T_SZ + t] = s;
}

// =====================================================================
// Kernel 3: per-head scores  S = mix(dot, cos, -dist)  -> attn region (raw)
// ip = qh @ kh^T : K=64.  Q/K tiles are f16 already -> async DMA to LDS.
// =====================================================================
__global__ __launch_bounds__(256) void score_kernel(
    const _Float16* __restrict__ qp, const _Float16* __restrict__ kp,
    const float* __restrict__ q2, const float* __restrict__ k2,
    const float* __restrict__ sw, float* __restrict__ attn)
{
    const int BM = 128, BN = 128;
    __shared__ _Float16 Aq[BM * HD];   // 16KB
    __shared__ _Float16 Bk[BN * HD];   // 16KB
    __shared__ float q2s[BM], k2s[BN];

    const int tid = threadIdx.x, lane = tid & 31, wid = tid >> 5;
    const int h = blockIdx.z;
    const int bm = blockIdx.y * BM, bn = blockIdx.x * BN;

    // softmax over the 3 score weights (tiny, per-thread)
    const float e0 = __expf(sw[0]), e1 = __expf(sw[1]), e2 = __expf(sw[2]);
    const float wden = 1.0f / (e0 + e1 + e2);
    const float w0 = e0 * wden, w1 = e1 * wden, w2 = e2 * wden;

    // async DMA: 1024 x 16B chunks per matrix, 4 per thread per matrix
#pragma unroll
    for (int it = 0; it < 4; it++) {
        const int i = tid + it * 256;
        const int r = i >> 3, c = (i & 7) * 8;
        async_copy_b128(Aq + r * HD + c,
                        qp + (size_t)(bm + r) * D_SZ + h * HD + c);
    }
#pragma unroll
    for (int it = 0; it < 4; it++) {
        const int i = tid + it * 256;
        const int r = i >> 3, c = (i & 7) * 8;
        async_copy_b128(Bk + r * HD + c,
                        kp + (size_t)(bn + r) * D_SZ + h * HD + c);
    }
    if (tid < BM) q2s[tid] = q2[h * T_SZ + bm + tid];
    else          k2s[tid - BM] = k2[h * T_SZ + bn + (tid - BM)];
    wait_async();
    __syncthreads();

    const int wm = (wid >> 1) * 32, wn = (wid & 1) * 64;
    v8f acc[2][4];
    for (int i = 0; i < 2; i++) for (int j = 0; j < 4; j++) acc[i][j] = zero_v8f();

#pragma unroll
    for (int s = 0; s < 2; s++) {              // K = 64 = 2 x 32
#pragma unroll
        for (int mi = 0; mi < 2; mi++) {
            const _Float16* ap = Aq + (wm + mi * 16 + (lane & 15)) * HD
                                    + s * 32 + (lane >> 4) * 8;
            v16h af = frag2(ap, ap + 16);
#pragma unroll
            for (int ni = 0; ni < 4; ni++) {
                const _Float16* bp = Bk + (wn + ni * 16 + (lane & 15)) * HD
                                        + s * 32 + (lane >> 4) * 8;
                v16h bf = frag2(bp, bp + 16);
                acc[mi][ni] = wmma16(af, bf, acc[mi][ni]);
            }
        }
    }

#pragma unroll
    for (int mi = 0; mi < 2; mi++)
#pragma unroll
        for (int ni = 0; ni < 4; ni++) {
            const int Mb = bm + wm + mi * 16 + (lane >> 4) * 8;
            const int Nl = wn + ni * 16 + (lane & 15);
            const int N  = bn + Nl;
            const float k2v = k2s[Nl];
            const float kn  = fmaxf(sqrtf(k2v), 1e-8f);
#pragma unroll
            for (int r = 0; r < 8; r++) {
                const float q2v = q2s[wm + mi * 16 + (lane >> 4) * 8 + r];
                const float qn  = fmaxf(sqrtf(q2v), 1e-8f);
                const float ip  = acc[mi][ni][r];
                const float dotv = ip * 0.125f;                 // 1/sqrt(64)
                const float cosv = ip / (qn * kn);
                const float d2   = q2v + k2v - 2.0f * ip;
                const float dist = -sqrtf(fmaxf(d2, 1e-12f));
                attn[((size_t)h * T_SZ + Mb + r) * T_SZ + N] =
                    w0 * dotv + w1 * cosv + w2 * dist;
            }
        }
}

// =====================================================================
// Kernel 4: in-place row softmax over attn (one block per row of 2048)
// =====================================================================
__global__ __launch_bounds__(256) void softmax_kernel(float* __restrict__ attn)
{
    __shared__ float red[8];
    const int tid = threadIdx.x, wid = tid >> 5;
    float* p = attn + (size_t)blockIdx.x * T_SZ;

    float v[8];
    float m = -3.0e38f;
#pragma unroll
    for (int i = 0; i < 8; i++) { v[i] = p[tid + i * 256]; m = fmaxf(m, v[i]); }
    for (int off = 16; off; off >>= 1) m = fmaxf(m, __shfl_xor(m, off, 32));
    if ((tid & 31) == 0) red[wid] = m;
    __syncthreads();
#pragma unroll
    for (int i = 0; i < 8; i++) m = fmaxf(m, red[i]);
    __syncthreads();

    float s = 0.0f;
#pragma unroll
    for (int i = 0; i < 8; i++) { v[i] = __expf(v[i] - m); s += v[i]; }
    for (int off = 16; off; off >>= 1) s += __shfl_xor(s, off, 32);
    if ((tid & 31) == 0) red[wid] = s;
    __syncthreads();
    float tot = 0.0f;
#pragma unroll
    for (int i = 0; i < 8; i++) tot += red[i];
    const float inv = 1.0f / tot;
#pragma unroll
    for (int i = 0; i < 8; i++) p[tid + i * 256] = v[i] * inv;
}

// =====================================================================
// Kernel 5: ctx = attn @ vh (per head), f16 output in head-major columns.
// M=T, N=64, K=T.  BM=128 BN=64 BK=64, wave = 32x32 (2x2 tiles).
// =====================================================================
__global__ __launch_bounds__(256) void ctx_kernel(
    const float* __restrict__ attn, const _Float16* __restrict__ vp,
    _Float16* __restrict__ ctx)
{
    const int BM = 128, BN = 64, BK = 64, LDB = BK + 8; // 144B rows (16B mult.)
    __shared__ _Float16 Aa[BM * BK];    // 16KB
    __shared__ _Float16 Bv[BN * LDB];   // 9KB, transposed [n][k]

    const int tid = threadIdx.x, lane = tid & 31, wid = tid >> 5;
    const int h = blockIdx.y, bm = blockIdx.x * BM;
    const float* arow = attn + (size_t)h * T_SZ * T_SZ;
    const int wm = (wid >> 1) * 32, wn = (wid & 1) * 32;

    v8f acc[2][2];
    for (int i = 0; i < 2; i++) for (int j = 0; j < 2; j++) acc[i][j] = zero_v8f();

    const int ar0 = tid >> 4, ac4 = (tid & 15) * 4;  // A staging coords

    for (int k0 = 0; k0 < T_SZ; k0 += BK) {
        __syncthreads();
        // stage A: 128x64 attn fp32 -> f16 (8 batched loads per thread)
        v4f ax[8];
#pragma unroll
        for (int it = 0; it < 8; it++)
            ax[it] = *(const v4f*)(arow + (size_t)(bm + ar0 + it * 16) * T_SZ + k0 + ac4);
        // stage B: vp rows (k) f16, 2 batched loads, transposed LDS store
        v8h bxv[2];
#pragma unroll
        for (int it = 0; it < 2; it++) {
            const int i = tid + it * 256;
            const int kk = i >> 3, c = (i & 7) * 8;
            bxv[it] = *(const v8h*)(vp + (size_t)(k0 + kk) * D_SZ + h * HD + c);
        }
#pragma unroll
        for (int it = 0; it < 8; it++) {
            _Float16* d = Aa + (ar0 + it * 16) * BK + ac4;
            d[0] = (_Float16)ax[it][0]; d[1] = (_Float16)ax[it][1];
            d[2] = (_Float16)ax[it][2]; d[3] = (_Float16)ax[it][3];
        }
#pragma unroll
        for (int it = 0; it < 2; it++) {
            const int i = tid + it * 256;
            const int kk = i >> 3, c = (i & 7) * 8;
#pragma unroll
            for (int j = 0; j < 8; j++) Bv[(c + j) * LDB + kk] = bxv[it][j];
        }
        if (k0 + BK < T_SZ)
            __builtin_prefetch(arow + (size_t)(bm + ar0) * T_SZ + k0 + BK, 0, 1);
        __syncthreads();

#pragma unroll
        for (int s = 0; s < 2; s++) {
#pragma unroll
            for (int mi = 0; mi < 2; mi++) {
                const _Float16* ap = Aa + (wm + mi * 16 + (lane & 15)) * BK
                                        + s * 32 + (lane >> 4) * 8;
                v16h af = frag2(ap, ap + 16);
#pragma unroll
                for (int ni = 0; ni < 2; ni++) {
                    const _Float16* bp = Bv + (wn + ni * 16 + (lane & 15)) * LDB
                                            + s * 32 + (lane >> 4) * 8;
                    v16h bf = frag2(bp, bp + 16);
                    acc[mi][ni] = wmma16(af, bf, acc[mi][ni]);
                }
            }
        }
    }
#pragma unroll
    for (int mi = 0; mi < 2; mi++)
#pragma unroll
        for (int ni = 0; ni < 2; ni++) {
            const int Mb = bm + wm + mi * 16 + (lane >> 4) * 8;
            const int N  = wn + ni * 16 + (lane & 15);
#pragma unroll
            for (int r = 0; r < 8; r++)
                ctx[(size_t)(Mb + r) * D_SZ + h * HD + N] = (_Float16)acc[mi][ni][r];
        }
}

// =====================================================================
// Kernel 6: out = ctx16 @ Wo + bo + q   (fp32 result into d_out)
// A tile is f16 already -> async DMA to LDS.
// =====================================================================
__global__ __launch_bounds__(256) void outproj_kernel(
    const _Float16* __restrict__ Xh, const float* __restrict__ W,
    const float* __restrict__ bias, const float* __restrict__ resid,
    float* __restrict__ out)
{
    const int BM = 128, BN = 128, BK = 32, LDB = BK + 8;
    __shared__ _Float16 As[BM * BK];
    __shared__ _Float16 Bs[BN * LDB];

    const int tid = threadIdx.x, lane = tid & 31, wid = tid >> 5;
    const int bm = blockIdx.y * BM, bn = blockIdx.x * BN;
    const int wm = (wid >> 1) * 32, wn = (wid & 1) * 64;

    v8f acc[2][4];
    for (int i = 0; i < 2; i++) for (int j = 0; j < 4; j++) acc[i][j] = zero_v8f();

    const int bk0 = tid >> 5, bc4 = (tid & 31) * 4;

    for (int k0 = 0; k0 < D_SZ; k0 += BK) {
        __syncthreads();
        // stage A (f16 -> f16): async DMA, 2 x 16B per thread
#pragma unroll
        for (int it = 0; it < 2; it++) {
            const int i = tid + it * 256;
            const int r = i >> 2, c = (i & 3) * 8;
            async_copy_b128(As + r * BK + c,
                            Xh + (size_t)(bm + r) * D_SZ + k0 + c);
        }
        // stage B (W f32 -> LDS [n][k] f16)
        v4f bx[4];
#pragma unroll
        for (int it = 0; it < 4; it++)
            bx[it] = *(const v4f*)(W + (size_t)(k0 + bk0 + it * 8) * D_SZ + bn + bc4);
#pragma unroll
        for (int it = 0; it < 4; it++) {
#pragma unroll
            for (int j = 0; j < 4; j++)
                Bs[(bc4 + j) * LDB + (bk0 + it * 8)] = (_Float16)bx[it][j];
        }
        wait_async();
        __syncthreads();
#pragma unroll
        for (int mi = 0; mi < 2; mi++) {
            const _Float16* ap = As + (wm + mi * 16 + (lane & 15)) * BK + (lane >> 4) * 8;
            v16h af = frag2(ap, ap + 16);
#pragma unroll
            for (int ni = 0; ni < 4; ni++) {
                const _Float16* bp = Bs + (wn + ni * 16 + (lane & 15)) * LDB + (lane >> 4) * 8;
                v16h bf = frag2(bp, bp + 16);
                acc[mi][ni] = wmma16(af, bf, acc[mi][ni]);
            }
        }
    }
#pragma unroll
    for (int mi = 0; mi < 2; mi++)
#pragma unroll
        for (int ni = 0; ni < 4; ni++) {
            const int Mb = bm + wm + mi * 16 + (lane >> 4) * 8;
            const int N  = bn + wn + ni * 16 + (lane & 15);
            const float bv = bias[N];
#pragma unroll
            for (int r = 0; r < 8; r++) {
                const size_t o = (size_t)(Mb + r) * D_SZ + N;
                out[o] = acc[mi][ni][r] + bv + resid[o];
            }
        }
}

// =====================================================================
// Kernel 7: in-place LayerNorm over rows of 1024 (one block per row)
// =====================================================================
__global__ __launch_bounds__(256) void ln_kernel(
    float* __restrict__ out, const float* __restrict__ gamma,
    const float* __restrict__ beta)
{
    __shared__ float red[8];
    const int tid = threadIdx.x, wid = tid >> 5;
    float* p = out + (size_t)blockIdx.x * D_SZ;

    float v[4]; float s = 0.0f;
#pragma unroll
    for (int i = 0; i < 4; i++) { v[i] = p[tid + i * 256]; s += v[i]; }
    for (int off = 16; off; off >>= 1) s += __shfl_xor(s, off, 32);
    if ((tid & 31) == 0) red[wid] = s;
    __syncthreads();
    float tot = 0.0f;
#pragma unroll
    for (int i = 0; i < 8; i++) tot += red[i];
    const float mu = tot * (1.0f / D_SZ);
    __syncthreads();

    float s2 = 0.0f;
#pragma unroll
    for (int i = 0; i < 4; i++) { float d = v[i] - mu; s2 += d * d; }
    for (int off = 16; off; off >>= 1) s2 += __shfl_xor(s2, off, 32);
    if ((tid & 31) == 0) red[wid] = s2;
    __syncthreads();
    float vtot = 0.0f;
#pragma unroll
    for (int i = 0; i < 8; i++) vtot += red[i];
    const float scale = rsqrtf(vtot * (1.0f / D_SZ) + 1e-5f);
#pragma unroll
    for (int i = 0; i < 4; i++) {
        const int c = tid + i * 256;
        p[c] = (v[i] - mu) * scale * gamma[c] + beta[c];
    }
}

// =====================================================================
extern "C" void kernel_launch(void* const* d_in, const int* in_sizes, int n_in,
                              void* d_out, int out_size, void* d_ws, size_t ws_size,
                              hipStream_t stream) {
    const float* q  = (const float*)d_in[0];
    const float* k  = (const float*)d_in[1];
    const float* v  = (const float*)d_in[2];
    const float* Wq = (const float*)d_in[3];
    const float* bq = (const float*)d_in[4];
    const float* Wk = (const float*)d_in[5];
    const float* bk = (const float*)d_in[6];
    const float* Wv = (const float*)d_in[7];
    const float* bv = (const float*)d_in[8];
    const float* Wo = (const float*)d_in[9];
    const float* bo = (const float*)d_in[10];
    const float* sw = (const float*)d_in[11];
    const float* lg = (const float*)d_in[12];
    const float* lb = (const float*)d_in[13];

    const size_t TD = (size_t)T_SZ * D_SZ;
    _Float16* qp16  = (_Float16*)d_ws;
    _Float16* kp16  = qp16 + TD;
    _Float16* vp16  = kp16 + TD;
    _Float16* ctx16 = vp16 + TD;
    float*    q2    = (float*)(ctx16 + TD);
    float*    k2    = q2 + (size_t)NH * T_SZ;

    float* outp = (float*)d_out;          // normed, T*D
    float* attn = outp + TD;              // attn, NH*T*T

    dim3 blk(256);
    dim3 gP(D_SZ / 128, T_SZ / 128);      // (8,16)
    proj_kernel<<<gP, blk, 0, stream>>>(q, Wq, bq, qp16);
    proj_kernel<<<gP, blk, 0, stream>>>(k, Wk, bk, kp16);
    proj_kernel<<<gP, blk, 0, stream>>>(v, Wv, bv, vp16);

    norm_kernel<<<(2 * T_SZ * NH) / 256, blk, 0, stream>>>(qp16, kp16, q2, k2);

    dim3 gS(T_SZ / 128, T_SZ / 128, NH);  // (16,16,16)
    score_kernel<<<gS, blk, 0, stream>>>(qp16, kp16, q2, k2, sw, attn);

    softmax_kernel<<<NH * T_SZ, blk, 0, stream>>>(attn);

    dim3 gC(T_SZ / 128, NH);
    ctx_kernel<<<gC, blk, 0, stream>>>(attn, vp16, ctx16);

    outproj_kernel<<<gP, blk, 0, stream>>>(ctx16, Wo, bo, q, outp);

    ln_kernel<<<T_SZ, blk, 0, stream>>>(outp, lg, lb);
}
